// Model_42064909697156
// MI455X (gfx1250) — matrix operand order repaired and checked by reading, not compiled, not run on hardware
//
#include <hip/hip_runtime.h>
#include <hip/hip_bf16.h>
#include <math.h>

typedef __bf16 bf16_t;
typedef __attribute__((ext_vector_type(16))) __bf16 v16bf;
typedef __attribute__((ext_vector_type(8)))  float  v8f;

constexpr int cN = 8192, cD = 4, cF = 256, cH = 512, cA = 64, cL = 512;
constexpr int cE  = cN * cD;   // 32768 edges
constexpr int c4H = 4 * cH;    // 2048 gate width

__device__ __forceinline__ float sigm(float x) { return 1.0f / (1.0f + __expf(-x)); }
__device__ __forceinline__ int imin(int a, int b) { return a < b ? a : b; }

// ---------------------------------------------------------------------------
// WMMA fragment loaders (ISA 7.12.2 layouts, wave32)
// A 16x32 bf16: lane<16 -> M=lane, K in {0..7}u{16..23}; lane>=16 -> M=lane-16,
// K in {8..15}u{24..31}.
// ---------------------------------------------------------------------------
__device__ __forceinline__ v16bf frag_a_bf16(const bf16_t* __restrict__ A, int ldk,
                                             int row0, int kk, int lane, int gather) {
  int m = lane & 15;
  int kofs = kk + ((lane < 16) ? 0 : 8);
  int r = row0 + m;
  if (gather) { int i = r >> 2, d = r & 3; r = imin((i + 1 + d) * cD + d, cE - 1); }
  const bf16_t* p0 = A + (size_t)r * ldk + kofs;
  v16bf f;
#pragma unroll
  for (int j = 0; j < 8; ++j) { f[j] = p0[j]; f[8 + j] = p0[16 + j]; }
  return f;
}

// A fragment sourced from two fp32 buffers (feat = [ef | eb]), converted to bf16
__device__ __forceinline__ v16bf frag_a_feat(const float* __restrict__ ef,
                                             const float* __restrict__ eb,
                                             int row0, int kk, int lane) {
  int m = lane & 15;
  int kofs = kk + ((lane < 16) ? 0 : 8);
  int r = row0 + m;
  const float* src = (kofs < cH) ? (ef + (size_t)r * cH + kofs)
                                 : (eb + (size_t)r * cH + (kofs - cH));
  v16bf f;
#pragma unroll
  for (int j = 0; j < 8; ++j) { f[j] = (bf16_t)src[j]; f[8 + j] = (bf16_t)src[16 + j]; }
  return f;
}

// B 32x16 bf16 with B[k][n] = W[n][k] (W row-major [Nout, K]):
// lane<16 -> n=lane, K in kk+{0..7}u{16..23} ; lane>=16 -> n=lane-16, K in kk+{8..15}u{24..31}
__device__ __forceinline__ v16bf frag_b_w(const bf16_t* __restrict__ W, int ldk,
                                          int n0, int kk, int lane) {
  int n = lane & 15;
  int kbase = kk + ((lane < 16) ? 0 : 8);
  const bf16_t* p = W + (size_t)(n0 + n) * ldk + kbase;
  v16bf f;
#pragma unroll
  for (int j = 0; j < 8; ++j) { f[j] = p[j]; f[8 + j] = p[16 + j]; }
  return f;
}

// ---------------------------------------------------------------------------
// Elementwise helpers
// ---------------------------------------------------------------------------
__global__ void f32_to_bf16_kernel(const float* __restrict__ src,
                                   bf16_t* __restrict__ dst, long n) {
  long i = (long)blockIdx.x * blockDim.x + threadIdx.x;
  long stride = (long)gridDim.x * blockDim.x;
  for (; i < n; i += stride) dst[i] = (bf16_t)src[i];
}

__global__ void zero_f32_kernel(float* __restrict__ p, long n) {
  long i = (long)blockIdx.x * blockDim.x + threadIdx.x;
  long stride = (long)gridDim.x * blockDim.x;
  for (; i < n; i += stride) p[i] = 0.0f;
}

// ---------------------------------------------------------------------------
// Input-projection GEMM: C[E,4H](bf16) = A[E,F](bf16) @ W[4H,F]^T, fp32 accum.
// Block = 8 waves; wave computes 16x64 tile; grid = (E/16, 4H/512).
// gather!=0 applies the backward-edge row map on A.
// ---------------------------------------------------------------------------
__global__ __launch_bounds__(256) void gemm_gates_kernel(
    const bf16_t* __restrict__ A, const bf16_t* __restrict__ W,
    bf16_t* __restrict__ C, int gather) {
  const int lane = threadIdx.x & 31;
  const int wave = threadIdx.x >> 5;
  const int m0 = blockIdx.x * 16;
  const int n0 = blockIdx.y * 512 + wave * 64;

  v8f acc[4] = {};
  for (int kk = 0; kk < cF; kk += 32) {
    v16bf a = frag_a_bf16(A, cF, m0, kk, lane, gather);
#pragma unroll
    for (int ns = 0; ns < 4; ++ns) {
      v16bf b = frag_b_w(W, cF, n0 + ns * 16, kk, lane);
      acc[ns] = __builtin_amdgcn_wmma_f32_16x16x32_bf16(
          false, a, false, b, (short)0, acc[ns], false, false);
    }
  }
  const int n = lane & 15;
  const int moff = (lane < 16) ? 0 : 8;
#pragma unroll
  for (int ns = 0; ns < 4; ++ns)
#pragma unroll
    for (int v = 0; v < 8; ++v)
      C[(size_t)(m0 + moff + v) * c4H + (n0 + ns * 16 + n)] = (bf16_t)acc[ns][v];
}

// ---------------------------------------------------------------------------
// Persistent sequential scan: one 1024-thread block per direction.
// Reuses u[j] = Whh @ h[j] across the 4 arcs that consume it (4x FLOP cut).
// 4-slot rings: at step i, slot (i-4)&3 == i&3 is dead once stage 1 has run,
// so mod-4 indexing is safe and keeps LDS at 60.5 KB.
// ---------------------------------------------------------------------------
__global__ __launch_bounds__(1024) void lstm_scan_kernel(
    const bf16_t* __restrict__ gates_f, const bf16_t* __restrict__ gates_b,
    const bf16_t* __restrict__ Whh_f,   const bf16_t* __restrict__ Whh_b,
    const float* __restrict__ b_f,      const float* __restrict__ b_b,
    const float* __restrict__ edges,
    const float* __restrict__ Wa, const float* __restrict__ ba,
    const float* __restrict__ Wao, const float* __restrict__ bao,
    float* __restrict__ ef, float* __restrict__ eb) {
  extern __shared__ float smem[];
  float* u_ring = smem;                    // 4 * 4H
  float* c_ring = u_ring + 4 * c4H;        // 4 * H
  float* harc   = c_ring + 4 * cH;         // D * H
  float* carc   = harc + cD * cH;          // D * H
  float* hcur   = carc + cD * cH;          // H
  float* zbuf   = hcur + cH;               // D * A
  float* red    = zbuf + cD * cA;          // 16

  const int fwd = (blockIdx.x == 0);
  const bf16_t* gates = fwd ? gates_f : gates_b;
  const bf16_t* Whh   = fwd ? Whh_f : Whh_b;
  const float*  bias  = fwd ? b_f : b_b;
  float*        out   = fwd ? ef : eb;
  const int tid = threadIdx.x;

  for (int s = 0; s < cN; ++s) {
    const int i = fwd ? s : (cN - 1 - s);

    // ---- stage 1: gates + LSTM cell for the 4 arcs --------------------
    for (int t = tid; t < cD * cH; t += 1024) {
      const int d = t >> 9, j = t & (cH - 1);
      const int p = fwd ? (i - 1 - d) : (i + 1 + d);
      const bool valid = fwd ? (p >= 0) : (p < cN);
      const bf16_t* gx = gates + (size_t)(i * cD + d) * c4H;
      float ui = 0.f, uff = 0.f, ug = 0.f, uo = 0.f, cp = 0.f;
      if (valid) {
        const float* ur = u_ring + (p & 3) * c4H;
        ui = ur[j]; uff = ur[cH + j]; ug = ur[2 * cH + j]; uo = ur[3 * cH + j];
        cp = c_ring[(p & 3) * cH + j];
      }
      const float gi = (float)gx[j]          + bias[j]          + ui;
      const float gf = (float)gx[cH + j]     + bias[cH + j]     + uff;
      const float gg = (float)gx[2 * cH + j] + bias[2 * cH + j] + ug;
      const float go = (float)gx[3 * cH + j] + bias[3 * cH + j] + uo;
      const float cc = sigm(gf) * cp + sigm(gi) * tanhf(gg);
      harc[t] = sigm(go) * tanhf(cc);
      carc[t] = cc;
    }
    // prefetch next step's gate block while we sync/compute
    if (s + 1 < cN) {
      const int inext = fwd ? (i + 1) : (i - 1);
      __builtin_prefetch(gates + (size_t)(inext * cD) * c4H + (size_t)tid * 8, 0, 0);
    }
    __syncthreads();

    // ---- stage 2: attention MLP, 4-way split dot products -------------
    // Scratch = this step's (dead) u_ring slot: stage 1 consumed all four
    // slots already; slot (i & 3) is rewritten only in stage 4.
    {
      float* zpart = u_ring + (i & 3) * c4H;  // need 1024 of 2048 floats
      // producer: t = (d*cA + a)*4 + seg, each sums a 128-wide segment
      for (int t = tid; t < cD * cA * 4; t += 1024) {
        const int seg = t & 3;
        const int a = (t >> 2) & (cA - 1);
        const int d = t >> 8;
        const float* wa = Wa + (size_t)a * (cH + 3) + seg * 128;
        const float* hd = harc + d * cH + seg * 128;
        float acc = 0.f;
        for (int k = 0; k < 128; k += 4)
          acc += wa[k] * hd[k] + wa[k + 1] * hd[k + 1] +
                 wa[k + 2] * hd[k + 2] + wa[k + 3] * hd[k + 3];
        zpart[t] = acc;
      }
      __syncthreads();
      for (int t = tid; t < cD * cA; t += 1024) {
        const int d = t / cA, a = t % cA;
        const float* wa = Wa + (size_t)a * (cH + 3);
        const int esrc = fwd ? (i * cD + d) : imin((i + 1 + d) * cD + d, cE - 1);
        const float* xr = edges + (size_t)esrc * cF;
        float acc = ba[a] + zpart[4 * t] + zpart[4 * t + 1] +
                    zpart[4 * t + 2] + zpart[4 * t + 3] +
                    wa[cH] * xr[0] + wa[cH + 1] * xr[1] + wa[cH + 2] * xr[2];
        zbuf[t] = fmaxf(acc, 0.f);
      }
    }
    __syncthreads();
    if (tid < cD) {
      const int d = tid;
      float acc = bao[0];
      for (int a = 0; a < cA; ++a) acc += Wao[a] * zbuf[d * cA + a];
      const int p = fwd ? (i - 1 - d) : (i + 1 + d);
      const bool valid = fwd ? (p >= 0) : (p < cN);
      red[d] = valid ? tanhf(acc) : -1e9f;
    }
    __syncthreads();
    if (tid == 0) {
      const float mx = fmaxf(fmaxf(red[0], red[1]), fmaxf(red[2], red[3]));
      float w0 = __expf(red[0] - mx), w1 = __expf(red[1] - mx);
      float w2 = __expf(red[2] - mx), w3 = __expf(red[3] - mx);
      const float inv = 1.0f / (w0 + w1 + w2 + w3);
      red[4] = w0 * inv; red[5] = w1 * inv; red[6] = w2 * inv; red[7] = w3 * inv;
    }
    __syncthreads();

    // ---- stage 3: attention-pool into node state, emit arc features ---
    if (tid < cH) {
      const int j = tid;
      float hn = 0.f, cn = 0.f;
#pragma unroll
      for (int d = 0; d < cD; ++d) {
        const int p = fwd ? (i - 1 - d) : (i + 1 + d);
        const float m = (fwd ? (p >= 0) : (p < cN)) ? 1.f : 0.f;
        const float w = red[4 + d] * m;
        hn += w * harc[d * cH + j];
        cn += w * carc[d * cH + j];
      }
      hcur[j] = hn;
      c_ring[(i & 3) * cH + j] = cn;
    }
    for (int t = tid; t < cD * cH; t += 1024) {
      const int d = t >> 9, j = t & (cH - 1);
      const int p = fwd ? (i - 1 - d) : (i + 1 + d);
      const bool valid = fwd ? (p >= 0) : (p < cN);
      if (fwd)            out[(size_t)(i * cD + d) * cH + j] = valid ? harc[t] : 0.f;
      else if (valid)     out[(size_t)((i + 1 + d) * cD + d) * cH + j] = harc[t];
    }
    __syncthreads();

    // ---- stage 4: u[i] = Whh @ h[i]  (shared by arcs of 4 future nodes)
    for (int r = tid; r < c4H; r += 1024) {
      const bf16_t* wr = Whh + (size_t)r * cH;
      float acc = 0.f;
      for (int k = 0; k < cH; k += 8)
        acc += (float)wr[k]     * hcur[k]     + (float)wr[k + 1] * hcur[k + 1] +
               (float)wr[k + 2] * hcur[k + 2] + (float)wr[k + 3] * hcur[k + 3] +
               (float)wr[k + 4] * hcur[k + 4] + (float)wr[k + 5] * hcur[k + 5] +
               (float)wr[k + 6] * hcur[k + 6] + (float)wr[k + 7] * hcur[k + 7];
      u_ring[(i & 3) * c4H + r] = acc;
    }
    __syncthreads();
  }
}

// ---------------------------------------------------------------------------
// Decoder: out[e] = Wdo . relu(Wd @ [ef[e]|eb[e]] + bd) + bdo, fused epilogue.
// Block = 8 waves covering all L=512 cols for a 16-row tile; LDS atomics reduce.
// ---------------------------------------------------------------------------
__global__ __launch_bounds__(256) void decoder_kernel(
    const float* __restrict__ ef, const float* __restrict__ eb,
    const bf16_t* __restrict__ Wd, const float* __restrict__ bd,
    const float* __restrict__ Wdo, const float* __restrict__ bdo,
    float* __restrict__ out) {
  __shared__ float part[16];
  const int lane = threadIdx.x & 31;
  const int wave = threadIdx.x >> 5;
  const int m0 = blockIdx.x * 16;
  const int n0 = wave * 64;
  if (threadIdx.x < 16) part[threadIdx.x] = 0.f;
  __syncthreads();

  v8f acc[4] = {};
  for (int kk = 0; kk < 2 * cH; kk += 32) {
    v16bf a = frag_a_feat(ef, eb, m0, kk, lane);
#pragma unroll
    for (int ns = 0; ns < 4; ++ns) {
      v16bf b = frag_b_w(Wd, 2 * cH, n0 + ns * 16, kk, lane);
      acc[ns] = __builtin_amdgcn_wmma_f32_16x16x32_bf16(
          false, a, false, b, (short)0, acc[ns], false, false);
    }
  }
  const int n = lane & 15;
  const int moff = (lane < 16) ? 0 : 8;
#pragma unroll
  for (int ns = 0; ns < 4; ++ns) {
    const int col = n0 + ns * 16 + n;
    const float wdo = Wdo[col];
    const float bdc = bd[col];
#pragma unroll
    for (int v = 0; v < 8; ++v) {
      const float hv = fmaxf(acc[ns][v] + bdc, 0.f);
      atomicAdd(&part[moff + v], hv * wdo);
    }
  }
  __syncthreads();
  if (threadIdx.x < 16) out[m0 + threadIdx.x] = part[threadIdx.x] + bdo[0];
}

// ---------------------------------------------------------------------------
extern "C" void kernel_launch(void* const* d_in, const int* in_sizes, int n_in,
                              void* d_out, int out_size, void* d_ws, size_t ws_size,
                              hipStream_t stream) {
  const float* edges = (const float*)d_in[0];
  const float* Wih_f = (const float*)d_in[1];
  const float* Whh_f = (const float*)d_in[2];
  const float* b_f   = (const float*)d_in[3];
  const float* Wih_b = (const float*)d_in[4];
  const float* Whh_b = (const float*)d_in[5];
  const float* b_b   = (const float*)d_in[6];
  const float* Wa    = (const float*)d_in[7];
  const float* ba    = (const float*)d_in[8];
  const float* Wao   = (const float*)d_in[9];
  const float* bao   = (const float*)d_in[10];
  const float* Wd    = (const float*)d_in[11];
  const float* bd    = (const float*)d_in[12];
  const float* Wdo   = (const float*)d_in[13];
  const float* bdo   = (const float*)d_in[14];
  float* out = (float*)d_out;
  (void)in_sizes; (void)n_in; (void)out_size; (void)ws_size;

  // Workspace carve-up (256B aligned)
  size_t off = 0;
  auto carve = [&](size_t bytes) {
    void* p = (char*)d_ws + off;
    off += (bytes + 255) & ~(size_t)255;
    return p;
  };
  bf16_t* edges_bf = (bf16_t*)carve((size_t)cE * cF * 2);
  bf16_t* Wihf_bf  = (bf16_t*)carve((size_t)c4H * cF * 2);
  bf16_t* Wihb_bf  = (bf16_t*)carve((size_t)c4H * cF * 2);
  bf16_t* Whhf_bf  = (bf16_t*)carve((size_t)c4H * cH * 2);
  bf16_t* Whhb_bf  = (bf16_t*)carve((size_t)c4H * cH * 2);
  bf16_t* Wd_bf    = (bf16_t*)carve((size_t)cL * 2 * cH * 2);
  bf16_t* gates_f  = (bf16_t*)carve((size_t)cE * c4H * 2);
  bf16_t* gates_b  = (bf16_t*)carve((size_t)cE * c4H * 2);
  float*  ef       = (float*)carve((size_t)cE * cH * 4);
  float*  eb       = (float*)carve((size_t)cE * cH * 4);

  // 1) fp32 -> bf16 conversions, zero the scatter target
  const dim3 cgrid(2048), cblk(256);
  f32_to_bf16_kernel<<<cgrid, cblk, 0, stream>>>(edges, edges_bf, (long)cE * cF);
  f32_to_bf16_kernel<<<cgrid, cblk, 0, stream>>>(Wih_f, Wihf_bf, (long)c4H * cF);
  f32_to_bf16_kernel<<<cgrid, cblk, 0, stream>>>(Wih_b, Wihb_bf, (long)c4H * cF);
  f32_to_bf16_kernel<<<cgrid, cblk, 0, stream>>>(Whh_f, Whhf_bf, (long)c4H * cH);
  f32_to_bf16_kernel<<<cgrid, cblk, 0, stream>>>(Whh_b, Whhb_bf, (long)c4H * cH);
  f32_to_bf16_kernel<<<cgrid, cblk, 0, stream>>>(Wd, Wd_bf, (long)cL * 2 * cH);
  zero_f32_kernel<<<cgrid, cblk, 0, stream>>>(eb, (long)cE * cH);

  // 2) input-projection GEMMs (WMMA bf16 -> fp32 -> bf16 gates)
  dim3 ggrid(cE / 16, c4H / 512);
  gemm_gates_kernel<<<ggrid, dim3(256), 0, stream>>>(edges_bf, Wihf_bf, gates_f, 0);
  gemm_gates_kernel<<<ggrid, dim3(256), 0, stream>>>(edges_bf, Wihb_bf, gates_b, 1);

  // 3) sequential bidirectional scan (2 persistent blocks, 1 per direction)
  const size_t scan_lds =
      (size_t)(4 * c4H + 4 * cH + 2 * cD * cH + cH + cD * cA + 16) * sizeof(float);
  lstm_scan_kernel<<<dim3(2), dim3(1024), scan_lds, stream>>>(
      gates_f, gates_b, Whhf_bf, Whhb_bf, b_f, b_b, edges,
      Wa, ba, Wao, bao, ef, eb);

  // 4) decoder GEMM with fused final dot-product
  decoder_kernel<<<dim3(cE / 16), dim3(256), 0, stream>>>(
      ef, eb, Wd_bf, bd, Wdo, bdo, out);
}
